// SynthModel_79834852098095
// MI455X (gfx1250) — matrix operand order repaired and checked by reading, not compiled, not run on hardware
//
#include <hip/hip_runtime.h>
#include <hip/hip_bf16.h>

typedef __attribute__((ext_vector_type(16))) _Float16 v16h;
typedef __attribute__((ext_vector_type(8)))  _Float16 h8;
typedef __attribute__((ext_vector_type(8)))  float    v8f;

#define BB 16
#define HH 64
#define NBANDS 65
#define FF 200
#define NSMP 96000
#define IRLEN 48000
#define FRLEN 480
#define NBINS 241
#define DRY_PAD 48016
#define DRYP_STRIDE 144064          // 48016 + 96000 + 48   (16B aligned rows)
#define IRREP_STRIDE 48160          // 48 front guard + window + tail guard
#define NTILEQ 4                    // output tiles (of 16 samples) per wave
#define NCHUNK 1503                 // K chunks of 32 taps (covers 4-tile union)

__device__ __forceinline__ float softplusf(float x) {
  return (x > 20.f) ? x : log1pf(__expf(x));
}

// ---------------- activations: amps[b,f,h] = softmax(hw)*softplus(gw)*mask ----
__global__ void k_activations(const float* __restrict__ hw,
                              const float* __restrict__ gw,
                              const float* __restrict__ f0,
                              float* __restrict__ amps) {
  int row = blockIdx.x;            // b*FF + f
  int h = threadIdx.x;             // 0..63
  __shared__ float sd[64];
  float x = hw[row * HH + h];
  sd[h] = x; __syncthreads();
  for (int off = 32; off > 0; off >>= 1) {
    if (h < off) sd[h] = fmaxf(sd[h], sd[h + off]);
    __syncthreads();
  }
  float mx = sd[0]; __syncthreads();
  float e = __expf(x - mx);
  sd[h] = e; __syncthreads();
  for (int off = 32; off > 0; off >>= 1) {
    if (h < off) sd[h] += sd[h + off];
    __syncthreads();
  }
  float sum = sd[0];
  float g = softplusf(gw[row]);
  float f0v = f0[row];
  float mask = (f0v * (float)(h + 1) < 24000.0f) ? 1.0f : 0.0f;
  amps[row * HH + h] = (e / sum) * g * mask;
}

// ---------------- f0 upsample chunk sums (480-sample chunks) ------------------
__global__ void k_chunksum(const float* __restrict__ f0, float* __restrict__ cs) {
  int blk = blockIdx.x;            // b*FF + c
  int b = blk / FF, c = blk % FF;
  int t = threadIdx.x;             // 0..511
  __shared__ float sd[512];
  float v = 0.f;
  if (t < FRLEN) {
    int n = c * FRLEN + t;
    double dpos = (double)n * (199.0 / 95999.0);
    int i0 = (int)dpos; if (i0 > FF - 1) i0 = FF - 1;
    int i1 = min(i0 + 1, FF - 1);
    float w = (float)(dpos - (double)i0);
    v = f0[b * FF + i0] * (1.f - w) + f0[b * FF + i1] * w;
  }
  sd[t] = v; __syncthreads();
  for (int off = 256; off > 0; off >>= 1) {
    if (t < off) sd[t] += sd[t + off];
    __syncthreads();
  }
  if (t == 0) cs[blk] = sd[0];
}

// ---------------- serial exclusive prefix over 200 chunks / batch -------------
__global__ void k_prefix(const float* __restrict__ cs, float* __restrict__ cp) {
  int b = threadIdx.x;
  if (b < BB) {
    float r = 0.f;
    for (int c = 0; c < FF; ++c) { cp[b * FF + c] = r; r += cs[b * FF + c]; }
  }
}

// ---------------- harmonic oscillator bank (fused scan + synth) ---------------
__global__ void k_harm(const float* __restrict__ f0,
                       const float* __restrict__ amps,
                       const float* __restrict__ cp,
                       float* __restrict__ dry) {
  int blk = blockIdx.x;            // b*FF + c
  int b = blk / FF, c = blk % FF;
  int t = threadIdx.x;             // 0..511
  __shared__ float sd[512];
  int n = c * FRLEN + t;
  double dpos = (double)n * (199.0 / 95999.0);
  int i0 = (int)dpos; if (i0 > FF - 1) i0 = FF - 1;
  int i1 = min(i0 + 1, FF - 1);
  float w = (float)(dpos - (double)i0);
  float f0v = 0.f;
  if (t < FRLEN) f0v = f0[b * FF + i0] * (1.f - w) + f0[b * FF + i1] * w;
  sd[t] = f0v; __syncthreads();
  for (int off = 1; off < 512; off <<= 1) {       // inclusive Hillis-Steele scan
    float add = (t >= off) ? sd[t - off] : 0.f;
    __syncthreads();
    sd[t] += add;
    __syncthreads();
  }
  if (t < FRLEN) {
    float cum = cp[b * FF + c] + sd[t];
    float rev = cum * (1.0f / 48000.0f);
    rev = rev - floorf(rev);                      // phase in revolutions [0,1)
    float ph = rev * 6.28318530718f;
    float s1, c1; __sincosf(ph, &s1, &c1);
    float sh = 0.f, ch = 1.f, acc = 0.f;
    const float* a0 = amps + (b * FF + i0) * HH;
    const float* a1 = amps + (b * FF + i1) * HH;
    #pragma unroll 8
    for (int h = 0; h < HH; ++h) {                // sin(k*ph) via rotation
      float ns = sh * c1 + ch * s1;
      float nc = ch * c1 - sh * s1;
      sh = ns; ch = nc;
      float a = a0[h] * (1.f - w) + a1[h] * w;
      acc = fmaf(a, sh, acc);
    }
    dry[b * NSMP + n] = acc;
  }
}

// ---------------- per-frame DFT noise filtering -------------------------------
__global__ void k_noise(const float* __restrict__ noise,
                        const float* __restrict__ naw,
                        float* __restrict__ dry) {
  int blk = blockIdx.x;            // b*FF + f
  int b = blk / FF, f = blk % FF;
  int t = threadIdx.x;             // 0..255
  __shared__ float fr[FRLEN];
  __shared__ float ct[FRLEN], st[FRLEN];
  __shared__ float zr[NBINS], zi[NBINS];
  for (int i = t; i < FRLEN; i += 256) {
    fr[i] = noise[b * NSMP + f * FRLEN + i];
    float ang = (float)i * (6.28318530718f / 480.f);
    ct[i] = __cosf(ang); st[i] = __sinf(ang);
  }
  __syncthreads();
  if (t < NBINS) {
    int k = t;
    float re = 0.f, im = 0.f;
    int m = 0;
    for (int s = 0; s < FRLEN; ++s) {            // twiddle idx = (k*s)%480
      re = fmaf(fr[s], ct[m], re);
      im = fmaf(-fr[s], st[m], im);
      m += k; if (m >= FRLEN) m -= FRLEN;
    }
    float bpos = (float)k * (64.f / 240.f);
    int b0 = (int)bpos; if (b0 > NBANDS - 1) b0 = NBANDS - 1;
    int b1 = min(b0 + 1, NBANDS - 1);
    float bw = bpos - (float)b0;
    float m0 = softplusf(naw[(b * FF + f) * NBANDS + b0]);
    float m1 = softplusf(naw[(b * FF + f) * NBANDS + b1]);
    float mag = m0 * (1.f - bw) + m1 * bw;
    zr[k] = re * mag; zi[k] = im * mag;
  }
  __syncthreads();
  for (int s = t; s < FRLEN; s += 256) {
    float acc = zr[0];
    int m = 0;
    for (int k = 1; k < 240; ++k) {
      m += s; if (m >= FRLEN) m -= FRLEN;
      acc += 2.f * (zr[k] * ct[m] - zi[k] * st[m]);
    }
    acc += (s & 1) ? -zr[240] : zr[240];
    dry[b * NSMP + f * FRLEN + s] += acc * (1.f / 480.f);
  }
}

// ---------------- pack dry -> padded f16 rows ---------------------------------
__global__ void k_pack_dry(const float* __restrict__ dry, _Float16* __restrict__ d16) {
  int idx = blockIdx.x * 256 + threadIdx.x;
  if (idx >= BB * DRYP_STRIDE) return;
  int b = idx / DRYP_STRIDE, p = idx % DRYP_STRIDE;
  int t = p - DRY_PAD;
  float v = (t >= 0 && t < NSMP) ? dry[b * NSMP + t] : 0.f;
  d16[idx] = (_Float16)v;
}

// ------- 16 phase-shifted reversed-IR replicas (aligned Toeplitz loads) -------
// irrep[j][U] = ir_wet[48080 - U + j]  (0 outside [1, IRLEN))
__global__ void k_pack_ir(const float* __restrict__ ir, _Float16* __restrict__ irrep) {
  int idx = blockIdx.x * 256 + threadIdx.x;
  if (idx >= 16 * IRREP_STRIDE) return;
  int j = idx / IRREP_STRIDE, u = idx % IRREP_STRIDE;
  int x = 48080 - u + j;                         // ir tap index
  float v = (x >= 1 && x < IRLEN) ? ir[x] : 0.f; // ir[0] forced to 0 (ir_wet)
  irrep[idx] = (_Float16)v;
}

// ---------------- WMMA Toeplitz reverb: out = dry + dry (x) ir ----------------
// One wave computes a 16(batch) x 64(sample) strip as 4 WMMA tiles sharing A.
__device__ __forceinline__ v16h load16(const _Float16* p, int second_off) {
  h8 lo = *(const h8*)(p);
  h8 hi = *(const h8*)(p + second_off);
  return __builtin_shufflevector(lo, hi, 0,1,2,3,4,5,6,7,8,9,10,11,12,13,14,15);
}

__global__ void k_reverb_wmma(const _Float16* __restrict__ d16,
                              const _Float16* __restrict__ irrep,
                              const float* __restrict__ dry,
                              float* __restrict__ out) {
  const int wave = threadIdx.x >> 5;
  const int lane = threadIdx.x & 31;
  const int grp = blockIdx.x * 8 + wave;         // strip id (uniform per wave)
  if (grp >= 1500) return;
  const int n0 = grp * (16 * NTILEQ);
  const int mj = lane & 15;                      // A row (batch) / B,D column
  const int hi = lane >> 4;

  // A: dry[batch=mj, t]; ISA 16x32 f16 layout: lane-half selects K groups
  const _Float16* aB = d16 + (size_t)mj * DRYP_STRIDE + n0 + hi * 8;
  // B: Toeplitz via per-column replica row; tile q window = base shifted -16q
  const _Float16* bB = irrep + (size_t)mj * IRREP_STRIDE + hi * 8;

  v8f acc0 = {}, acc1 = {}, acc2 = {}, acc3 = {};
  for (int kk = 0; kk < NCHUNK; ++kk) {
    const int s = kk * 32;
    v16h a = load16(aB + s, 16);                 // shared across the 4 tiles
    v16h b0 = load16(bB + s + 64, 16);
    v16h b1 = load16(bB + s + 48, 16);
    v16h b2 = load16(bB + s + 32, 16);
    v16h b3 = load16(bB + s + 16, 16);
    acc0 = __builtin_amdgcn_wmma_f32_16x16x32_f16(false, a, false, b0, (short)0, acc0, false, false);
    acc1 = __builtin_amdgcn_wmma_f32_16x16x32_f16(false, a, false, b1, (short)0, acc1, false, false);
    acc2 = __builtin_amdgcn_wmma_f32_16x16x32_f16(false, a, false, b2, (short)0, acc2, false, false);
    acc3 = __builtin_amdgcn_wmma_f32_16x16x32_f16(false, a, false, b3, (short)0, acc3, false, false);
  }

  // D layout: lanes 0-15 -> M=0..7 in VGPR r, lanes 16-31 -> M=8..15
  #pragma unroll
  for (int r = 0; r < 8; ++r) {
    const int bi = hi * 8 + r;                   // batch
    const int base = bi * NSMP + n0 + mj;
    out[base +  0] = dry[base +  0] + acc0[r];
    out[base + 16] = dry[base + 16] + acc1[r];
    out[base + 32] = dry[base + 32] + acc2[r];
    out[base + 48] = dry[base + 48] + acc3[r];
  }
}

extern "C" void kernel_launch(void* const* d_in, const int* in_sizes, int n_in,
                              void* d_out, int out_size, void* d_ws, size_t ws_size,
                              hipStream_t stream) {
  const float* hw    = (const float*)d_in[0];   // [B,F,H]
  const float* nw    = (const float*)d_in[1];   // [B,F,NB]
  const float* gw    = (const float*)d_in[2];   // [B,F,1]
  const float* f0    = (const float*)d_in[3];   // [B,F,1]
  const float* ir    = (const float*)d_in[4];   // [IR_LEN]
  const float* noise = (const float*)d_in[5];   // [B,NS]
  float* out = (float*)d_out;

  char* ws = (char*)d_ws;
  float*    amps  = (float*)(ws + 0);           //   819,200 B
  float*    csum  = (float*)(ws + 819200);      //    12,800 B
  float*    cpre  = (float*)(ws + 832000);      //    12,800 B
  float*    dry   = (float*)(ws + 844800);      // 6,144,000 B
  _Float16* d16   = (_Float16*)(ws + 6988800);  // 4,610,048 B
  _Float16* irrep = (_Float16*)(ws + 11598848); // 1,541,120 B  (total ~12.5 MB)

  k_activations<<<BB * FF, 64, 0, stream>>>(hw, gw, f0, amps);
  k_chunksum<<<BB * FF, 512, 0, stream>>>(f0, csum);
  k_prefix<<<1, 32, 0, stream>>>(csum, cpre);
  k_harm<<<BB * FF, 512, 0, stream>>>(f0, amps, cpre, dry);
  k_noise<<<BB * FF, 256, 0, stream>>>(noise, nw, dry);
  k_pack_ir<<<(16 * IRREP_STRIDE + 255) / 256, 256, 0, stream>>>(ir, irrep);
  k_pack_dry<<<(BB * DRYP_STRIDE + 255) / 256, 256, 0, stream>>>(dry, d16);
  k_reverb_wmma<<<188, 256, 0, stream>>>(d16, irrep, dry, out);
}
